// MoveScorer_39333310496986
// MI455X (gfx1250) — matrix operand order repaired and checked by reading, not compiled, not run on hardware
//
#include <hip/hip_runtime.h>
#include <hip/hip_bf16.h>
#include <math.h>

typedef unsigned int   u32;
typedef unsigned short u16;
typedef __attribute__((ext_vector_type(16))) __bf16 v16bf;
typedef __attribute__((ext_vector_type(8)))  float  v8f;

// ---------- helpers ----------
__device__ __forceinline__ u16 f2bf(float f) {
    u32 x = __float_as_uint(f);
    u32 r = x + 0x7FFFu + ((x >> 16) & 1u);   // round-to-nearest-even
    return (u16)(r >> 16);
}
__device__ __forceinline__ float bf2f(u16 b) {
    return __uint_as_float(((u32)b) << 16);
}
// pack two f32 -> one dword of two bf16 (lo = a, hi = b), RNE
__device__ __forceinline__ u32 pack2bf(float a, float b) {
    u32 xa = __float_as_uint(a);
    u32 xb = __float_as_uint(b);
    xa = xa + 0x7FFFu + ((xa >> 16) & 1u);
    xb = xb + 0x7FFFu + ((xb >> 16) & 1u);
    return (xa >> 16) | (xb & 0xFFFF0000u);
}

union AFrag { v16bf v; uint4 q[2]; u32 d[8]; u16 s[16]; };

// Load B fragment from LDS-staged copy: layout [kb][nb][lane][8 dwords]
__device__ __forceinline__ v16bf load_bfrag(const u32* frag, int kb, int nb, int lane) {
    AFrag u;
    const u32* p = frag + ((kb * 4 + nb) * 32 + lane) * 8;
    u.q[0] = *(const uint4*)p;
    u.q[1] = *(const uint4*)(p + 4);
    return u.v;
}

// ---------- fp32 -> bf16 convert ----------
__global__ void cvt_bf16_kernel(const float* __restrict__ x, u16* __restrict__ y, size_t n) {
    size_t i = (size_t)blockIdx.x * blockDim.x + threadIdx.x;
    if (i < n) y[i] = f2bf(x[i]);
}

// ---------- pack weight [K,64] fp32 row-major into WMMA B-fragments ----------
__global__ void pack_w_kernel(const float* __restrict__ W, u32* __restrict__ frag, int K) {
    int total = (K / 32) * 4 * 32 * 8;
    int tid = blockIdx.x * blockDim.x + threadIdx.x;
    if (tid >= total) return;
    int j    = tid & 7;
    int lane = (tid >> 3) & 31;
    int nb   = (tid >> 8) & 3;
    int kb   = tid >> 10;
    int kbase = kb * 32 + 8 * (lane >> 4) + 2 * (j & 3) + 16 * (j >> 2);
    int n     = nb * 16 + (lane & 15);
    frag[tid] = pack2bf(W[(size_t)kbase * 64 + n], W[(size_t)(kbase + 1) * 64 + n]);
}

// ---------- degree / inverse-degree ----------
__global__ void deg_kernel(const long long* __restrict__ dst, float* __restrict__ deg, int E) {
    int i = blockIdx.x * blockDim.x + threadIdx.x;
    if (i < E) atomicAdd(&deg[(int)dst[i]], 1.0f);
}
__global__ void inv_kernel(const float* __restrict__ deg, float* __restrict__ inv, int N) {
    int i = blockIdx.x * blockDim.x + threadIdx.x;
    if (i < N) { float d = deg[i]; inv[i] = d > 0.f ? 1.f / d : 0.f; }
}

// ---------- GEMM: out[rows,64] = relu(A[rows,K] @ W + b), A bf16, out bf16 ----------
template <int K>
__global__ void gemm_bias_relu_kernel(const u16* __restrict__ A, const u32* __restrict__ frag,
                                      const float* __restrict__ bias, u16* __restrict__ out, int nrows) {
    constexpr int KB = K / 32;
    __shared__ u32 sfrag[KB * 1024];
    for (int i = threadIdx.x; i < KB * 256; i += 256)
        ((uint4*)sfrag)[i] = ((const uint4*)frag)[i];
    __syncthreads();

    int wave = threadIdx.x >> 5, lane = threadIdx.x & 31;
    int row0 = blockIdx.x * 128 + wave * 16;
    if (row0 >= nrows) return;
    int rload = row0 + (lane & 15); if (rload >= nrows) rload = nrows - 1;
    int halfo = (lane >> 4) * 8;
    const u16* arow = A + (size_t)rload * K + halfo;

    // preload all A fragments (one clause, latency overlapped with WMMA chain)
    AFrag ua[KB];
#pragma unroll
    for (int kb = 0; kb < KB; ++kb) {
        ua[kb].q[0] = *(const uint4*)(arow + kb * 32);
        ua[kb].q[1] = *(const uint4*)(arow + kb * 32 + 16);
    }

    v8f acc[4] = {};
#pragma unroll
    for (int kb = 0; kb < KB; ++kb) {
#pragma unroll
        for (int nb = 0; nb < 4; ++nb) {
            v16bf b = load_bfrag(sfrag, kb, nb, lane);
            acc[nb] = __builtin_amdgcn_wmma_f32_16x16x32_bf16(false, ua[kb].v, false, b,
                                                              (short)0, acc[nb], false, false);
        }
    }
    int nsub = lane & 15, mhalf = (lane >> 4) * 8;
#pragma unroll
    for (int nb = 0; nb < 4; ++nb) {
        float bv = bias[nb * 16 + nsub];
#pragma unroll
        for (int r = 0; r < 8; ++r) {
            int row = row0 + mhalf + r;
            if (row < nrows) {
                float v = acc[nb][r] + bv;
                v = v > 0.f ? v : 0.f;
                out[(size_t)row * 64 + nb * 16 + nsub] = f2bf(v);
            }
        }
    }
}

// ---------- SAGE mean-aggregation: agg[dst] += h[src] (wave per edge, f32 atomics) ----------
__global__ void sage_agg_kernel(const u16* __restrict__ h, const long long* __restrict__ src,
                                const long long* __restrict__ dst, float* __restrict__ agg, int E) {
    int tid  = blockIdx.x * blockDim.x + threadIdx.x;
    int e    = tid >> 5;
    int lane = tid & 31;
    if (e >= E) return;
    int s = (int)src[e], d = (int)dst[e];
    int c = lane * 2;
    u32 hv = *(const u32*)(h + (size_t)s * 64 + c);
    float f0 = bf2f((u16)(hv & 0xFFFF));
    float f1 = bf2f((u16)(hv >> 16));
    atomicAdd(&agg[(size_t)d * 64 + c],     f0);
    atomicAdd(&agg[(size_t)d * 64 + c + 1], f1);
}

// ---------- SAGE output: out = relu( (agg*inv) @ Wl + h @ Wr + b ), K = 64 ----------
__global__ void sage_gemm_kernel(const float* __restrict__ agg, const float* __restrict__ inv,
                                 const u16* __restrict__ h,
                                 const u32* __restrict__ fragL, const u32* __restrict__ fragR,
                                 const float* __restrict__ bias, u16* __restrict__ out, int nrows) {
    __shared__ u32 sfragL[2048];
    __shared__ u32 sfragR[2048];
    for (int i = threadIdx.x; i < 512; i += 256) {
        ((uint4*)sfragL)[i] = ((const uint4*)fragL)[i];
        ((uint4*)sfragR)[i] = ((const uint4*)fragR)[i];
    }
    __syncthreads();

    int wave = threadIdx.x >> 5, lane = threadIdx.x & 31;
    int row0 = blockIdx.x * 128 + wave * 16;
    if (row0 >= nrows) return;
    int rload = row0 + (lane & 15); if (rload >= nrows) rload = nrows - 1;
    int halfo = (lane >> 4) * 8;

    // preload A1 (agg, f32) and A2 (h, bf16) for both k-steps up front.
    // NOTE: keep float4 values as SSA values (.x/.y access only, no address
    // taken) so all 8 b128 loads issue as one clause.
    float invd = inv[rload];
    const float* ap = agg + (size_t)rload * 64 + halfo;
    float4 f[2][4];
#pragma unroll
    for (int kb = 0; kb < 2; ++kb) {
        f[kb][0] = *(const float4*)(ap + kb * 32);
        f[kb][1] = *(const float4*)(ap + kb * 32 + 4);
        f[kb][2] = *(const float4*)(ap + kb * 32 + 16);
        f[kb][3] = *(const float4*)(ap + kb * 32 + 20);
    }
    const u16* hrow = h + (size_t)rload * 64 + halfo;
    AFrag uh[2];
#pragma unroll
    for (int kb = 0; kb < 2; ++kb) {
        uh[kb].q[0] = *(const uint4*)(hrow + kb * 32);
        uh[kb].q[1] = *(const uint4*)(hrow + kb * 32 + 16);
    }

    v8f acc[4] = {};
#pragma unroll
    for (int kb = 0; kb < 2; ++kb) {
        AFrag ua;
        ua.d[0] = pack2bf(f[kb][0].x * invd, f[kb][0].y * invd);
        ua.d[1] = pack2bf(f[kb][0].z * invd, f[kb][0].w * invd);
        ua.d[2] = pack2bf(f[kb][1].x * invd, f[kb][1].y * invd);
        ua.d[3] = pack2bf(f[kb][1].z * invd, f[kb][1].w * invd);
        ua.d[4] = pack2bf(f[kb][2].x * invd, f[kb][2].y * invd);
        ua.d[5] = pack2bf(f[kb][2].z * invd, f[kb][2].w * invd);
        ua.d[6] = pack2bf(f[kb][3].x * invd, f[kb][3].y * invd);
        ua.d[7] = pack2bf(f[kb][3].z * invd, f[kb][3].w * invd);
#pragma unroll
        for (int nb = 0; nb < 4; ++nb) {
            v16bf b = load_bfrag(sfragL, kb, nb, lane);
            acc[nb] = __builtin_amdgcn_wmma_f32_16x16x32_bf16(false, ua.v, false, b,
                                                              (short)0, acc[nb], false, false);
        }
    }
#pragma unroll
    for (int kb = 0; kb < 2; ++kb) {
#pragma unroll
        for (int nb = 0; nb < 4; ++nb) {
            v16bf b = load_bfrag(sfragR, kb, nb, lane);
            acc[nb] = __builtin_amdgcn_wmma_f32_16x16x32_bf16(false, uh[kb].v, false, b,
                                                              (short)0, acc[nb], false, false);
        }
    }
    int nsub = lane & 15, mhalf = (lane >> 4) * 8;
#pragma unroll
    for (int nb = 0; nb < 4; ++nb) {
        float bv = bias[nb * 16 + nsub];
#pragma unroll
        for (int r = 0; r < 8; ++r) {
            int row = row0 + mhalf + r;
            if (row < nrows) {
                float v = acc[nb][r] + bv;
                v = v > 0.f ? v : 0.f;
                out[(size_t)row * 64 + nb * 16 + nsub] = f2bf(v);
            }
        }
    }
}

// ---------- edge MLP: tanh( relu([h[s],h[d]] @ W1 + b1) @ w2 + b2 ), wave = 16 edges ----------
__global__ void mlp_kernel(const u16* __restrict__ h, const long long* __restrict__ me,
                           const u32* __restrict__ fragW1, const float* __restrict__ b1,
                           const float* __restrict__ w2, const float* __restrict__ b2,
                           float* __restrict__ out, int M) {
    __shared__ u32 sfrag[4096];
    for (int i = threadIdx.x; i < 1024; i += 256)
        ((uint4*)sfrag)[i] = ((const uint4*)fragW1)[i];
    __syncthreads();

    int wave = threadIdx.x >> 5, lane = threadIdx.x & 31;
    int e0 = blockIdx.x * 128 + wave * 16;
    if (e0 >= M) return;
    int eL = e0 + (lane & 15); if (eL >= M) eL = M - 1;
    long long sN = me[eL];
    long long dN = me[(size_t)M + eL];
    const u16* srow = h + (size_t)sN * 64;
    const u16* drow = h + (size_t)dN * 64;
    int halfo = (lane >> 4) * 8;

    // preload all 4 A fragments (src halves then dst halves) as one clause
    AFrag ua[4];
#pragma unroll
    for (int kb = 0; kb < 4; ++kb) {
        const u16* base = (kb < 2 ? srow : drow) + (kb & 1) * 32 + halfo;
        ua[kb].q[0] = *(const uint4*)base;
        ua[kb].q[1] = *(const uint4*)(base + 16);
    }

    v8f acc[4] = {};
#pragma unroll
    for (int kb = 0; kb < 4; ++kb) {
#pragma unroll
        for (int nb = 0; nb < 4; ++nb) {
            v16bf b = load_bfrag(sfrag, kb, nb, lane);
            acc[nb] = __builtin_amdgcn_wmma_f32_16x16x32_bf16(false, ua[kb].v, false, b,
                                                              (short)0, acc[nb], false, false);
        }
    }
    // layer 2: logit[m] = sum_n relu(z[m,n]+b1[n]) * w2[n]
    int nsub = lane & 15;
    float p[8];
#pragma unroll
    for (int r = 0; r < 8; ++r) p[r] = 0.f;
#pragma unroll
    for (int nb = 0; nb < 4; ++nb) {
        float bb = b1[nb * 16 + nsub];
        float ww = w2[nb * 16 + nsub];
#pragma unroll
        for (int r = 0; r < 8; ++r) {
            float z = acc[nb][r] + bb;
            z = z > 0.f ? z : 0.f;
            p[r] += z * ww;
        }
    }
#pragma unroll
    for (int mask = 1; mask <= 8; mask <<= 1) {    // reduce within 16-lane halves
#pragma unroll
        for (int r = 0; r < 8; ++r) p[r] += __shfl_xor(p[r], mask, 32);
    }
    float b2v = b2[0];
    int half = lane >> 4, r = lane & 15;
    if (r < 8) {
        float logit = 0.f;
#pragma unroll
        for (int i = 0; i < 8; ++i) if (r == i) logit = p[i];
        int e = e0 + half * 8 + r;
        if (e < M) out[e] = tanhf(logit + b2v);
    }
}

// ---------- launch ----------
extern "C" void kernel_launch(void* const* d_in, const int* in_sizes, int n_in,
                              void* d_out, int out_size, void* d_ws, size_t ws_size,
                              hipStream_t stream) {
    const float*     x      = (const float*)d_in[0];
    const long long* ei     = (const long long*)d_in[1];
    const long long* mei    = (const long long*)d_in[2];
    const float*     lin1_w = (const float*)d_in[3];
    const float*     lin1_b = (const float*)d_in[4];
    const float*     lin2_w = (const float*)d_in[5];
    const float*     lin2_b = (const float*)d_in[6];
    const float*     c1wl   = (const float*)d_in[7];
    const float*     c1bl   = (const float*)d_in[8];
    const float*     c1wr   = (const float*)d_in[9];
    const float*     c2wl   = (const float*)d_in[10];
    const float*     c2bl   = (const float*)d_in[11];
    const float*     c2wr   = (const float*)d_in[12];
    const float*     mlp_w1 = (const float*)d_in[13];
    const float*     mlp_b1 = (const float*)d_in[14];
    const float*     mlp_w2 = (const float*)d_in[15];
    const float*     mlp_b2 = (const float*)d_in[16];

    const int N = in_sizes[0] / 128;
    const int E = in_sizes[1] / 2;
    const int M = in_sizes[2] / 2;

    char* ws = (char*)d_ws;
    size_t off = 0;
    auto carve = [&](size_t bytes) -> char* {
        char* p = ws + off;
        off = (off + bytes + 255) & ~(size_t)255;
        return p;
    };
    u16*  xbf  = (u16*)carve((size_t)N * 128 * 2);
    u16*  hA   = (u16*)carve((size_t)N * 64 * 2);
    u16*  hB   = (u16*)carve((size_t)N * 64 * 2);
    float* agg = (float*)carve((size_t)N * 64 * 4);
    float* deg = (float*)carve((size_t)N * 4);
    float* invd= (float*)carve((size_t)N * 4);
    u32* fL1  = (u32*)carve(4096 * 4);   // K=128
    u32* fL2  = (u32*)carve(2048 * 4);   // K=64
    u32* fC1L = (u32*)carve(2048 * 4);
    u32* fC1R = (u32*)carve(2048 * 4);
    u32* fC2L = (u32*)carve(2048 * 4);
    u32* fC2R = (u32*)carve(2048 * 4);
    u32* fM1  = (u32*)carve(4096 * 4);   // K=128

    hipMemsetAsync(deg, 0, (size_t)N * 4, stream);
    hipMemsetAsync(agg, 0, (size_t)N * 64 * 4, stream);

    {
        size_t n = (size_t)N * 128;
        cvt_bf16_kernel<<<(unsigned)((n + 255) / 256), 256, 0, stream>>>(x, xbf, n);
    }
    pack_w_kernel<<<16, 256, 0, stream>>>(lin1_w, fL1, 128);
    pack_w_kernel<<<8,  256, 0, stream>>>(lin2_w, fL2, 64);
    pack_w_kernel<<<8,  256, 0, stream>>>(c1wl,  fC1L, 64);
    pack_w_kernel<<<8,  256, 0, stream>>>(c1wr,  fC1R, 64);
    pack_w_kernel<<<8,  256, 0, stream>>>(c2wl,  fC2L, 64);
    pack_w_kernel<<<8,  256, 0, stream>>>(c2wr,  fC2R, 64);
    pack_w_kernel<<<16, 256, 0, stream>>>(mlp_w1, fM1, 128);

    deg_kernel<<<(E + 255) / 256, 256, 0, stream>>>(ei + E, deg, E);
    inv_kernel<<<(N + 255) / 256, 256, 0, stream>>>(deg, invd, N);

    int gblk = (N + 127) / 128;
    // h = relu(x @ lin1 + b)
    gemm_bias_relu_kernel<128><<<gblk, 256, 0, stream>>>(xbf, fL1, lin1_b, hA, N);
    // conv1
    sage_agg_kernel<<<(E + 7) / 8, 256, 0, stream>>>(hA, ei, ei + E, agg, E);
    sage_gemm_kernel<<<gblk, 256, 0, stream>>>(agg, invd, hA, fC1L, fC1R, c1bl, hB, N);
    // h = relu(h @ lin2 + b)
    gemm_bias_relu_kernel<64><<<gblk, 256, 0, stream>>>(hB, fL2, lin2_b, hA, N);
    // conv2
    hipMemsetAsync(agg, 0, (size_t)N * 64 * 4, stream);
    sage_agg_kernel<<<(E + 7) / 8, 256, 0, stream>>>(hA, ei, ei + E, agg, E);
    sage_gemm_kernel<<<gblk, 256, 0, stream>>>(agg, invd, hA, fC2L, fC2R, c2bl, hB, N);
    // edge MLP + tanh
    mlp_kernel<<<(M + 127) / 128, 256, 0, stream>>>(hB, mei, fM1, mlp_b1, mlp_w2, mlp_b2,
                                                    (float*)d_out, M);
}
